// TransformerNetwork_81535659147447
// MI455X (gfx1250) — compile-verified, matrix operand order checked by reading, not run on hardware
//
#include <hip/hip_runtime.h>
#include <math.h>

typedef _Float16 f16;
typedef __attribute__((ext_vector_type(16))) _Float16 v16h;
typedef __attribute__((ext_vector_type(8)))  _Float16 v8h;
typedef __attribute__((ext_vector_type(8)))  float    v8f;
typedef __attribute__((ext_vector_type(4)))  int      v4i;

#define B_    4
#define L_    2048
#define D_    1024
#define H_    4
#define NL_   2
#define HD_   256
#define FF_   4096
#define MTOT  (B_*L_)      // 8192 token rows
#define EPS_  1e-5f

// ---------------------------------------------------------------------------
// Async global->LDS staging (CDNA5 GLOBAL_LOAD_ASYNC_TO_LDS_B128, ASYNCcnt),
// guarded so the code still compiles if the toolchain lacks the builtins.
// Signature (probe-confirmed): (global v4i*, local v4i*, imm, imm).
// ---------------------------------------------------------------------------
#if __has_builtin(__builtin_amdgcn_global_load_async_to_lds_b128) && \
    __has_builtin(__builtin_amdgcn_s_wait_asynccnt)
#define HAVE_ASYNC_LDS 1
#else
#define HAVE_ASYNC_LDS 0
#endif

typedef __attribute__((address_space(1))) v4i gv4i_t;
typedef __attribute__((address_space(3))) v4i lv4i_t;

__device__ __forceinline__ void copy_g2l_b128(f16* dst_lds, const f16* src_glb) {
#if HAVE_ASYNC_LDS
  __builtin_amdgcn_global_load_async_to_lds_b128(
      (gv4i_t*)(const_cast<f16*>(src_glb)),
      (lv4i_t*)dst_lds, 0, 0);
#else
  *(v8h*)dst_lds = *(const v8h*)src_glb;
#endif
}

// Wait until at most `n` async copies (per wave) are still outstanding.
// Async loads retire in order, so lag-N waiting gives double buffering.
__device__ __forceinline__ void async_wait_upto(int n) {
#if HAVE_ASYNC_LDS
  if (n == 0)       __builtin_amdgcn_s_wait_asynccnt(0);
  else if (n == 5)  __builtin_amdgcn_s_wait_asynccnt(5);
  else              __builtin_amdgcn_s_wait_asynccnt(0);
#endif
}

// ---------------------------------------------------------------------------
// WMMA fragment helpers (16x16x32 f16, wave32).
// A-fragment (16 rows x 32 K): lane<16 -> row=lane, halves [0..7]=K 0..7,
// [8..15]=K16..23 ; lane>=16 -> row=lane-16, K bases 8 and 24.
// B-fragment uses the identical per-lane layout with "row" = N.
// ---------------------------------------------------------------------------
__device__ __forceinline__ v16h make_frag(v8h lo, v8h hi) {
  return __builtin_shufflevector(lo, hi, 0,1,2,3,4,5,6,7,8,9,10,11,12,13,14,15);
}

__device__ __forceinline__ v16h load_frag(const f16* tile, int rowBase, int ld,
                                          int kChunkBase, int lane) {
  int r  = rowBase + (lane & 15);
  int kb = kChunkBase + ((lane < 16) ? 0 : 8);
  const f16* p = tile + (size_t)r * ld + kb;
  v8h lo = *(const v8h*)(p);
  v8h hi = *(const v8h*)(p + 16);
  return make_frag(lo, hi);
}

// ---------------------------------------------------------------------------
// f32 -> f16 conversion (grid-stride)
// ---------------------------------------------------------------------------
__global__ void cvt_f32_f16_kernel(const float* __restrict__ in, f16* __restrict__ out,
                                   size_t n) {
  size_t i = (size_t)blockIdx.x * blockDim.x + threadIdx.x;
  size_t stride = (size_t)gridDim.x * blockDim.x;
  for (; i < n; i += stride) out[i] = (f16)in[i];
}

// ---------------------------------------------------------------------------
// LayerNorm over D_=1024: one 256-thread block per row.
// ---------------------------------------------------------------------------
template<bool OUT_F16>
__global__ void __launch_bounds__(256)
ln_kernel(const float* __restrict__ x, const float* __restrict__ g,
          const float* __restrict__ bb, void* __restrict__ out) {
  __shared__ float red[256];
  int row = blockIdx.x;
  const float* xr = x + (size_t)row * D_;
  float v[4];
  float s = 0.f;
  #pragma unroll
  for (int i = 0; i < 4; i++) { v[i] = xr[threadIdx.x + i*256]; s += v[i]; }
  red[threadIdx.x] = s; __syncthreads();
  for (int o = 128; o > 0; o >>= 1) {
    if ((int)threadIdx.x < o) red[threadIdx.x] += red[threadIdx.x + o];
    __syncthreads();
  }
  float mu = red[0] * (1.0f / D_);
  __syncthreads();
  float s2 = 0.f;
  #pragma unroll
  for (int i = 0; i < 4; i++) { float d = v[i] - mu; s2 += d * d; }
  red[threadIdx.x] = s2; __syncthreads();
  for (int o = 128; o > 0; o >>= 1) {
    if ((int)threadIdx.x < o) red[threadIdx.x] += red[threadIdx.x + o];
    __syncthreads();
  }
  float rstd = rsqrtf(red[0] * (1.0f / D_) + EPS_);
  #pragma unroll
  for (int i = 0; i < 4; i++) {
    int c = threadIdx.x + i*256;
    float y = (v[i] - mu) * rstd * g[c] + bb[c];
    if (OUT_F16) ((f16*)out)[(size_t)row*D_ + c] = (f16)y;
    else         ((float*)out)[(size_t)row*D_ + c] = y;
  }
}

// ---------------------------------------------------------------------------
// Generic WMMA GEMM:  out[M,N] = A[M,K](f16) * W[N,K](f16)^T + bias
//                     (+ residual) (+ exact GELU); out f32 or f16.
// Block = 256 threads = 8 waves, 4(M) x 2(N); block tile 64x256;
// each wave computes one 16x128 strip (8 accumulators = 64 VGPRs).
// K-step 32, double-buffered LDS staged with async global->LDS copies:
// chunk i+1's copies fly while chunk i is consumed (s_wait_asynccnt lag 5 =
// per-thread async ops per chunk).
// ---------------------------------------------------------------------------
template<bool OUT_F16, bool RES, bool GELU>
__global__ void __launch_bounds__(256)
gemm_kernel(const f16* __restrict__ A, const f16* __restrict__ W,
            const float* __restrict__ bias, const float* __restrict__ resid,
            void* __restrict__ out, int M, int N, int K) {
  __shared__ f16 As[2][64*32];
  __shared__ f16 Ws[2][256*32];
  int tid  = threadIdx.x;
  int lane = tid & 31, wave = tid >> 5;
  int waveM = wave >> 1, waveN = wave & 1;
  int blockN0 = blockIdx.x * 256, blockM0 = blockIdx.y * 64;

  const int nchunks = K >> 5;   // K is a multiple of 32

  // issue one chunk's async copies into buffer `buf` (5 b128 ops per thread)
  auto issue = [&](int chunk, int buf) {
    int k0 = chunk * 32;
    {  // A tile: 64x32 halves = 256 b128, 1 per thread
      int r = tid >> 2, c8 = (tid & 3) * 8;
      copy_g2l_b128(&As[buf][r*32 + c8], &A[(size_t)(blockM0 + r)*K + k0 + c8]);
    }
    #pragma unroll
    for (int i = 0; i < 4; i++) {  // W tile: 256x32 halves = 1024 b128, 4 per thread
      int id = tid + i*256;
      int r = id >> 2, c8 = (id & 3) * 8;
      copy_g2l_b128(&Ws[buf][r*32 + c8], &W[(size_t)(blockN0 + r)*K + k0 + c8]);
    }
  };

  v8f acc[8] = {};
  issue(0, 0);
  for (int c = 0; c < nchunks; c++) {
    int buf = c & 1;
    if (c + 1 < nchunks) issue(c + 1, buf ^ 1);       // prefetch next chunk
    async_wait_upto((c + 1 < nchunks) ? 5 : 0);       // chunk c landed
    __syncthreads();
    v16h af = load_frag(As[buf], waveM*16, 32, 0, lane);
    #pragma unroll
    for (int t = 0; t < 8; t++) {
      v16h bf = load_frag(Ws[buf], waveN*128 + t*16, 32, 0, lane);
      acc[t] = __builtin_amdgcn_wmma_f32_16x16x32_f16(false, af, false, bf,
                                                      (short)0, acc[t], false, false);
    }
    __syncthreads();   // all reads of buf done before it is overwritten
  }

  int mHalf = (lane < 16) ? 0 : 8;
  int n0 = lane & 15;
  #pragma unroll
  for (int t = 0; t < 8; t++) {
    int gcol = blockN0 + waveN*128 + t*16 + n0;
    float bv = bias[gcol];
    #pragma unroll
    for (int j = 0; j < 8; j++) {
      int grow = blockM0 + waveM*16 + mHalf + j;
      float v = acc[t][j] + bv;
      if (GELU) v = 0.5f * v * (1.0f + erff(v * 0.70710678118f));
      size_t idx = (size_t)grow * N + gcol;
      if (RES) v += resid[idx];
      if (OUT_F16) ((f16*)out)[idx] = (f16)v;
      else         ((float*)out)[idx] = v;
    }
  }
}

// ---------------------------------------------------------------------------
// Flash attention (non-causal), f16 in / f16 out, f32 online-softmax state.
// Block = 256 threads = 8 waves; q-block 64. Waves pair up per 16-row q-tile:
//   wave (qt, kh): S = Q Kt^T for key-half kh (2 of 4 key tiles, K=256 in 8
//   chunks) -> shared S buffer;  softmax row-wise by the kh==0 wave;
//   O(16 x 128) += P V for d-half kh (8 accumulators per wave).
// Q/K tiles are staged row-major via async copies; V staged transposed (d,key)
// so P*V's B-fragments are plain row-major loads.
// qkv layout: row stride 3D, q at col h*HD, k at D+h*HD, v at 2D+h*HD.
// ---------------------------------------------------------------------------
__global__ void __launch_bounds__(256, 1)
attn_kernel(const f16* __restrict__ qkv, f16* __restrict__ o) {
  extern __shared__ char smem[];
  f16*   Qt = (f16*)smem;                 // [64][256]
  f16*   Kt = Qt + 64*256;                // [64][256]
  f16*   Vt = Kt + 64*256;                // [256][64]  (transposed V)
  float* Sb = (float*)(Vt + 256*64);      // [4][16][64]
  f16*   Pb = (f16*)(Sb + 4*16*64);       // [4][16][64]
  float* Al = (float*)(Pb + 4*16*64);     // [4][16] alpha
  float* Li = Al + 4*16;                  // [4][16] 1/l

  int tid = threadIdx.x, lane = tid & 31, wave = tid >> 5;
  int qt = wave >> 1;        // q-tile 0..3
  int dh = wave & 1;         // key-half for S, d-half for O
  int b = blockIdx.z, h = blockIdx.y;
  int q0 = blockIdx.x * 64;
  const size_t rs = 3*D_;
  const f16* qb = qkv + (size_t)b*L_*rs +        (size_t)h*HD_;
  const f16* kp = qkv + (size_t)b*L_*rs + D_   + (size_t)h*HD_;
  const f16* vp = qkv + (size_t)b*L_*rs + 2*D_ + (size_t)h*HD_;

  int m     = lane & 15;
  int mHalf = (lane < 16) ? 0 : 8;

  // stage Q tile once: 64x256 halves = 2048 b128, 8 per thread
  #pragma unroll
  for (int i = 0; i < 8; i++) {
    int id = tid + i*256;
    int r = id >> 5, c8 = (id & 31) * 8;
    copy_g2l_b128(&Qt[r*256 + c8], &qb[(size_t)(q0 + r)*rs + c8]);
  }

  v8f oacc[8] = {};
  float rm = -INFINITY, rl = 0.0f;
  const float scale = 0.0625f;   // 1/sqrt(256)

  float* Sw = Sb + qt*16*64;
  f16*   Pw = Pb + qt*16*64;
  float* Aw = Al + qt*16;
  float* Lw = Li + qt*16;

  for (int kb0 = 0; kb0 < L_; kb0 += 64) {
    // K tile (async, row-major)
    #pragma unroll
    for (int i = 0; i < 8; i++) {
      int id = tid + i*256;
      int r = id >> 5, c8 = (id & 31) * 8;
      copy_g2l_b128(&Kt[r*256 + c8], &kp[(size_t)(kb0 + r)*rs + c8]);
    }
    // V tile, transposed into (d, key)
    #pragma unroll
    for (int i = 0; i < 8; i++) {
      int id = tid + i*256;
      int r = id >> 5, c8 = (id & 31) * 8;
      v8h vv = *(const v8h*)&vp[(size_t)(kb0 + r)*rs + c8];
      #pragma unroll
      for (int j = 0; j < 8; j++) Vt[(c8 + j)*64 + r] = vv[j];
    }
    async_wait_upto(0);
    __syncthreads();

    // S(16 x 32) for this wave's key-half
    v8f s[2] = {};
    #pragma unroll
    for (int c = 0; c < 8; c++) {
      v16h af = load_frag(Qt, qt*16, 256, c*32, lane);
      #pragma unroll
      for (int t = 0; t < 2; t++) {
        v16h bf = load_frag(Kt, dh*32 + t*16, 256, c*32, lane);
        s[t] = __builtin_amdgcn_wmma_f32_16x16x32_f16(false, af, false, bf,
                                                      (short)0, s[t], false, false);
      }
    }
    // spill S to the q-tile's shared buffer, (q,key) row-major f32
    #pragma unroll
    for (int t = 0; t < 2; t++)
      #pragma unroll
      for (int j = 0; j < 8; j++)
        Sw[(mHalf + j)*64 + dh*32 + t*16 + m] = s[t][j];
    __syncthreads();

    // online softmax: kh==0 wave, lane r < 16 owns row r of the q-tile
    if (dh == 0 && lane < 16) {
      const float* srow = Sw + lane*64;
      float mx = rm;
      for (int c2 = 0; c2 < 64; c2++) mx = fmaxf(mx, srow[c2] * scale);
      float alpha = __expf(rm - mx);
      float sum = 0.0f;
      f16* prow = Pw + lane*64;
      for (int c2 = 0; c2 < 64; c2++) {
        float p = __expf(srow[c2] * scale - mx);
        prow[c2] = (f16)p;
        sum += p;
      }
      rl = rl * alpha + sum;
      rm = mx;
      Aw[lane] = alpha;
    }
    __syncthreads();

    // rescale O accumulators by per-row alpha
    float a8[8];
    #pragma unroll
    for (int j = 0; j < 8; j++) a8[j] = Aw[mHalf + j];
    #pragma unroll
    for (int dt = 0; dt < 8; dt++)
      #pragma unroll
      for (int j = 0; j < 8; j++) oacc[dt][j] *= a8[j];

    // O(16 x 128) += P V for this wave's d-half
    v16h pf0 = load_frag(Pw, 0, 64, 0,  lane);
    v16h pf1 = load_frag(Pw, 0, 64, 32, lane);
    #pragma unroll
    for (int dt = 0; dt < 8; dt++) {
      v16h bf0 = load_frag(Vt, dh*128 + dt*16, 64, 0, lane);
      oacc[dt] = __builtin_amdgcn_wmma_f32_16x16x32_f16(false, pf0, false, bf0,
                                                        (short)0, oacc[dt], false, false);
      v16h bf1 = load_frag(Vt, dh*128 + dt*16, 64, 32, lane);
      oacc[dt] = __builtin_amdgcn_wmma_f32_16x16x32_f16(false, pf1, false, bf1,
                                                        (short)0, oacc[dt], false, false);
    }
    __syncthreads();
  }

  if (dh == 0 && lane < 16) Lw[lane] = 1.0f / rl;
  __syncthreads();
  float l8[8];
  #pragma unroll
  for (int j = 0; j < 8; j++) l8[j] = Lw[mHalf + j];

  #pragma unroll
  for (int dt = 0; dt < 8; dt++) {
    #pragma unroll
    for (int j = 0; j < 8; j++) {
      int grow = q0 + qt*16 + mHalf + j;
      int gcol = h*HD_ + dh*128 + dt*16 + m;
      o[((size_t)b*L_ + grow)*D_ + gcol] = (f16)(oacc[dt][j] * l8[j]);
    }
  }
}

// ---------------------------------------------------------------------------
// Host orchestration
// ---------------------------------------------------------------------------
static inline size_t align256(size_t x) { return (x + 255) & ~(size_t)255; }

extern "C" void kernel_launch(void* const* d_in, const int* in_sizes, int n_in,
                              void* d_out, int out_size, void* d_ws, size_t ws_size,
                              hipStream_t stream) {
  (void)in_sizes; (void)n_in; (void)out_size; (void)ws_size;
  const float* x     = (const float*)d_in[0];
  const float* w3d   = (const float*)d_in[1];
  const float* b3d   = (const float*)d_in[2];
  const float* ln1_g = (const float*)d_in[3];
  const float* ln1_b = (const float*)d_in[4];
  const float* wqkv  = (const float*)d_in[5];
  const float* bqkv  = (const float*)d_in[6];
  const float* wproj = (const float*)d_in[7];
  const float* bproj = (const float*)d_in[8];
  const float* ln2_g = (const float*)d_in[9];
  const float* ln2_b = (const float*)d_in[10];
  const float* w1    = (const float*)d_in[11];
  const float* b1    = (const float*)d_in[12];
  const float* w2    = (const float*)d_in[13];
  const float* b2    = (const float*)d_in[14];
  const float* lnf_g = (const float*)d_in[15];
  const float* lnf_b = (const float*)d_in[16];

  // workspace carve-up (~292 MB)
  char* ws = (char*)d_ws;
  size_t off = 0;
  auto carve = [&](size_t bytes) -> void* { void* p = ws + off; off += align256(bytes); return p; };
  f16*   xh     = (f16*)  carve((size_t)MTOT*3*D_*sizeof(f16));
  f16*   w3dh   = (f16*)  carve((size_t)D_*3*D_*sizeof(f16));
  f16*   wqkvh  = (f16*)  carve((size_t)NL_*3*D_*D_*sizeof(f16));
  f16*   wprojh = (f16*)  carve((size_t)NL_*D_*D_*sizeof(f16));
  f16*   w1h    = (f16*)  carve((size_t)NL_*FF_*D_*sizeof(f16));
  f16*   w2h    = (f16*)  carve((size_t)NL_*D_*FF_*sizeof(f16));
  float* hbuf   = (float*)carve((size_t)MTOT*D_*sizeof(float));
  f16*   hn     = (f16*)  carve((size_t)MTOT*D_*sizeof(f16));
  f16*   qkvh   = (f16*)  carve((size_t)MTOT*3*D_*sizeof(f16));
  f16*   oh     = (f16*)  carve((size_t)MTOT*D_*sizeof(f16));
  f16*   fh     = (f16*)  carve((size_t)MTOT*FF_*sizeof(f16));

  // 1) f32 -> f16 conversions
  auto cvt = [&](const float* src, f16* dst, size_t n) {
    cvt_f32_f16_kernel<<<2048, 256, 0, stream>>>(src, dst, n);
  };
  cvt(x,     xh,     (size_t)MTOT*3*D_);
  cvt(w3d,   w3dh,   (size_t)D_*3*D_);
  cvt(wqkv,  wqkvh,  (size_t)NL_*3*D_*D_);
  cvt(wproj, wprojh, (size_t)NL_*D_*D_);
  cvt(w1,    w1h,    (size_t)NL_*FF_*D_);
  cvt(w2,    w2h,    (size_t)NL_*D_*FF_);

  dim3 blk(256);
  // 2) embedding projection: h = x * w3d^T + b3d   (M=8192, N=1024, K=3072)
  gemm_kernel<false,false,false><<<dim3(D_/256, MTOT/64), blk, 0, stream>>>(
      xh, w3dh, b3d, nullptr, hbuf, MTOT, D_, 3*D_);

  const size_t attnSmem = (size_t)(2*64*256 + 256*64)*sizeof(f16)  // Qt, Kt, Vt
                        + (size_t)(4*16*64)*sizeof(float)           // Sb
                        + (size_t)(4*16*64)*sizeof(f16)             // Pb
                        + (size_t)(2*4*16)*sizeof(float);           // Al, Li

  for (int i = 0; i < NL_; i++) {
    // ln1 -> hn (f16)
    ln_kernel<true><<<MTOT, blk, 0, stream>>>(hbuf, ln1_g + (size_t)i*D_,
                                              ln1_b + (size_t)i*D_, hn);
    // qkv = hn * wqkv^T + bqkv   (N=3D, f16 out)
    gemm_kernel<true,false,false><<<dim3(3*D_/256, MTOT/64), blk, 0, stream>>>(
        hn, wqkvh + (size_t)i*3*D_*D_, bqkv + (size_t)i*3*D_, nullptr,
        qkvh, MTOT, 3*D_, D_);
    // flash attention -> oh (f16)
    attn_kernel<<<dim3(L_/64, H_, B_), dim3(256), attnSmem, stream>>>(qkvh, oh);
    // h = h + oh * wproj^T + bproj   (in-place residual)
    gemm_kernel<false,true,false><<<dim3(D_/256, MTOT/64), blk, 0, stream>>>(
        oh, wprojh + (size_t)i*D_*D_, bproj + (size_t)i*D_, hbuf,
        hbuf, MTOT, D_, D_);
    // ln2 -> hn (f16)
    ln_kernel<true><<<MTOT, blk, 0, stream>>>(hbuf, ln2_g + (size_t)i*D_,
                                              ln2_b + (size_t)i*D_, hn);
    // f = gelu(hn * w1^T + b1)   (N=FF, f16 out)
    gemm_kernel<true,false,true><<<dim3(FF_/256, MTOT/64), blk, 0, stream>>>(
        hn, w1h + (size_t)i*FF_*D_, b1 + (size_t)i*FF_, nullptr,
        fh, MTOT, FF_, D_);
    // h = h + f * w2^T + b2   (K=FF, in-place residual)
    gemm_kernel<false,true,false><<<dim3(D_/256, MTOT/64), blk, 0, stream>>>(
        fh, w2h + (size_t)i*D_*FF_, b2 + (size_t)i*D_, hbuf,
        hbuf, MTOT, D_, FF_);
  }

  // final layernorm -> d_out (f32)
  ln_kernel<false><<<MTOT, blk, 0, stream>>>(hbuf, lnf_g, lnf_b, (float*)d_out);
}
